// LinearAttention_26070451487310
// MI455X (gfx1250) — compile-verified, matrix operand order checked by reading, not run on hardware
//
#include <hip/hip_runtime.h>
#include <hip/hip_bf16.h>

// ---------------------------------------------------------------------------
// Types for CDNA5 WMMA (wave32, 16x16x32 bf16 -> f32) and TDM descriptors
// ---------------------------------------------------------------------------
typedef __attribute__((ext_vector_type(8)))  __bf16 v8bf;
typedef __attribute__((ext_vector_type(16))) __bf16 v16bf;
typedef __attribute__((ext_vector_type(8)))  float  v8f;
typedef __attribute__((ext_vector_type(4)))  unsigned int u32x4;
typedef __attribute__((ext_vector_type(8)))  int i32x8;
typedef __attribute__((ext_vector_type(4)))  int i32x4;

__device__ inline float bf16_to_f32(unsigned short u) {
    unsigned int b = ((unsigned int)u) << 16;
    float f;
    __builtin_memcpy(&f, &b, 4);
    return f;
}
__device__ inline unsigned short f32_to_bf16(float f) {
    unsigned int b;
    __builtin_memcpy(&b, &f, 4);
    unsigned int r = b + 0x7FFFu + ((b >> 16) & 1u);  // round-to-nearest-even
    return (unsigned short)(r >> 16);
}

// Fragment loader for 16-bit 16x32 operand tiles from K-contiguous memory
// (row stride `ld` elements, 16-byte aligned rows).
// CDNA5 A layout: lanes 0-15 hold M=lane, K={0..7,16..23};
// lanes 16-31 hold M=lane-16, K={8..15,24..31}. B operand fed from B^T.
__device__ inline v16bf load_frag(const unsigned short* base, int ld) {
    int lane = threadIdx.x & 31;
    int row  = lane & 15;
    int koff = (lane >> 4) << 3;  // 0 or 8
    const unsigned short* p = base + (size_t)row * ld + koff;
    v8bf lo = *reinterpret_cast<const v8bf*>(p);        // K = koff..koff+7
    v8bf hi = *reinterpret_cast<const v8bf*>(p + 16);   // K = koff+16..koff+23
    return __builtin_shufflevector(lo, hi, 0, 1, 2, 3, 4, 5, 6, 7,
                                   8, 9, 10, 11, 12, 13, 14, 15);
}

__device__ inline v8f wmma_bf16(v16bf a, v16bf b, v8f c) {
    return __builtin_amdgcn_wmma_f32_16x16x32_bf16(
        /*neg_a=*/false, a, /*neg_b=*/false, b,
        /*c_mod=*/(short)0, c, /*reuse_a=*/false, /*reuse_b=*/false);
}

// ---------------------------------------------------------------------------
// Tensor Data Mover: async 2-D tile load (rows x 32 bf16) global -> LDS.
// D# per CDNA5 ISA ch.8: group0 = {count/flags, lds_addr, global_addr, type=2},
// group1 = {data_size=2B, tensor_dim0/1, tile_dim0=32, tile_dim1, stride}.
// Values are wave-uniform (SGPR descriptor groups). Tracked by TENSORcnt.
// ---------------------------------------------------------------------------
__device__ inline unsigned lds_addr_u32(const void* p) {
    // low 32 bits of the generic address of an LDS object = LDS byte offset
    return (unsigned)(size_t)p;
}

__device__ inline void tdm_load_tile_2d(const unsigned short* gbase,
                                        unsigned lds_byte_addr,
                                        unsigned dim0_rem,   // elems in a row remaining
                                        unsigned dim1_rem,   // rows remaining
                                        unsigned tile_rows,  // tile_dim1
                                        unsigned row_pitch)  // elems between rows
{
    unsigned long long ga = (unsigned long long)(size_t)gbase;  // byte address
    u32x4 g0 = {};
    g0[0] = 1u;                                     // count=1, user mode
    g0[1] = lds_byte_addr;                          // lds_addr
    g0[2] = (unsigned)(ga & 0xFFFFFFFFu);           // global_addr[95:64]
    g0[3] = (unsigned)((ga >> 32) & 0x01FFFFFFu) | (2u << 30);  // addr hi | type=2

    i32x8 g1 = {};
    g1[0] = (int)(1u << 16);                        // data_size=1 -> 2 bytes/elem
    g1[1] = (int)((dim0_rem & 0xFFFFu) << 16);      // tensor_dim0[15:0]
    g1[2] = (int)((dim0_rem >> 16) | ((dim1_rem & 0xFFFFu) << 16));  // d0 hi | d1 lo
    g1[3] = (int)((dim1_rem >> 16) | (32u << 16));  // d1 hi | tile_dim0 = 32
    g1[4] = (int)(tile_rows & 0xFFFFu);             // tile_dim1 | tile_dim2=0
    g1[5] = (int)row_pitch;                         // tensor_dim0_stride[31:0]
    g1[6] = 0;
    g1[7] = 0;

    i32x4 gz4 = {};
#if __clang_major__ >= 23
    i32x8 gz8 = {};
    __builtin_amdgcn_tensor_load_to_lds(g0, g1, gz4, gz4, gz8, 0);
#else
    __builtin_amdgcn_tensor_load_to_lds(g0, g1, gz4, gz4, 0);
#endif
}

// ---------------------------------------------------------------------------
// Elementwise converters
// ---------------------------------------------------------------------------
__global__ void cvt_f32_to_bf16(const float* __restrict__ in,
                                unsigned short* __restrict__ out, size_t n) {
    size_t i = (size_t)blockIdx.x * blockDim.x + threadIdx.x;
    size_t stride = (size_t)gridDim.x * blockDim.x;
    for (; i < n; i += stride) out[i] = f32_to_bf16(in[i]);
}

// W [1024][1024] fp32 row-major -> WT [n][k] bf16 (transposed)
__global__ void cvt_transpose_w(const float* __restrict__ W,
                                unsigned short* __restrict__ WT) {
    size_t i = (size_t)blockIdx.x * blockDim.x + threadIdx.x;  // 1M threads
    int k = (int)(i >> 10);
    int n = (int)(i & 1023);
    WT[(size_t)n * 1024 + k] = f32_to_bf16(W[i]);
}

// ---------------------------------------------------------------------------
// Main bf16 GEMM:  C[M][1024] = A[M][1024] @ W + bias
// A row-major bf16, W given as BT[n][k] bf16 (pre-transposed).
// Tiles staged via TDM tensor_load_to_lds, double-buffered so the DMA of
// tile i+1 overlaps WMMA on tile i. MODE 0: none, 1: elu(x)+1.
// ---------------------------------------------------------------------------
template <int MODE, bool F32OUT>
__global__ __launch_bounds__(256) void gemm_bf16_kernel(
    const unsigned short* __restrict__ A,
    const unsigned short* __restrict__ BT,
    const float* __restrict__ bias,
    void* __restrict__ Cout, int Mrows) {
    constexpr int N = 1024, K = 1024, NK = K / 32;
    __shared__ alignas(16) unsigned short As[2][128 * 32];
    __shared__ alignas(16) unsigned short Bs[2][128 * 32];

    const int m0 = blockIdx.y * 128;
    const int n0 = blockIdx.x * 128;
    const int t  = threadIdx.x;
    const int w  = t >> 5;
    const int wm = w >> 1;   // 0..3  -> 32 rows each
    const int wn = w & 1;    // 0..1  -> 64 cols each

    const unsigned short* Abase = A + (size_t)m0 * K;
    const unsigned short* Bbase = BT + (size_t)n0 * K;
    const unsigned mRem = (unsigned)(Mrows - m0);
    const unsigned nRem = (unsigned)(N - n0);

    const v8f vz = {};
    v8f acc[2][4];
#pragma unroll
    for (int i = 0; i < 2; i++)
#pragma unroll
        for (int j = 0; j < 4; j++) acc[i][j] = vz;

    // Prologue: kick off DMA for tile 0
    if (w == 0) {
        tdm_load_tile_2d(Abase, lds_addr_u32(&As[0][0]), K, mRem, 128, K);
        tdm_load_tile_2d(Bbase, lds_addr_u32(&Bs[0][0]), K, nRem, 128, K);
    }

    for (int i = 0; i < NK; i++) {
        const int cur = i & 1;
        if (w == 0) {
            if (i + 1 < NK) {
                const int k1 = (i + 1) * 32;
                const int nb = (i + 1) & 1;
                tdm_load_tile_2d(Abase + k1, lds_addr_u32(&As[nb][0]),
                                 (unsigned)(K - k1), mRem, 128, K);
                tdm_load_tile_2d(Bbase + k1, lds_addr_u32(&Bs[nb][0]),
                                 (unsigned)(K - k1), nRem, 128, K);
                __builtin_amdgcn_s_wait_tensorcnt(2);  // tile i pair complete
            } else {
                __builtin_amdgcn_s_wait_tensorcnt(0);  // drain
            }
        }
        __syncthreads();  // tile i visible to all waves

        v16bf af[2], bfr[4];
#pragma unroll
        for (int ii = 0; ii < 2; ii++)
            af[ii] = load_frag(&As[cur][(wm * 32 + ii * 16) * 32], 32);
#pragma unroll
        for (int j = 0; j < 4; j++)
            bfr[j] = load_frag(&Bs[cur][(wn * 64 + j * 16) * 32], 32);
#pragma unroll
        for (int ii = 0; ii < 2; ii++)
#pragma unroll
            for (int j = 0; j < 4; j++)
                acc[ii][j] = wmma_bf16(af[ii], bfr[j], acc[ii][j]);

        __syncthreads();  // done with buffer `cur` before TDM reuses it
    }

    // Epilogue. C layout: lane holds col = lane&15; VGPR r -> row r + 8*(lane>>4).
    const int lane = t & 31;
    const int colL = lane & 15;
    const int rhi  = (lane >> 4) * 8;
#pragma unroll
    for (int i = 0; i < 2; i++) {
#pragma unroll
        for (int j = 0; j < 4; j++) {
            int col  = n0 + wn * 64 + j * 16 + colL;
            float bv = bias[col];
#pragma unroll
            for (int r = 0; r < 8; r++) {
                int row   = m0 + wm * 32 + i * 16 + rhi + r;
                float val = acc[i][j][r] + bv;
                if (MODE == 1) val = (val > 0.0f) ? (val + 1.0f) : __expf(val);
                if (F32OUT)
                    ((float*)Cout)[(size_t)row * N + col] = val;
                else
                    ((unsigned short*)Cout)[(size_t)row * N + col] = f32_to_bf16(val);
            }
        }
    }
}

// ---------------------------------------------------------------------------
// kv[b,h,d,m] = sum_s k[b,s,h,d] * v[b,s,h,m]   (64x64 per head, WMMA-reduced)
// One block per (b,h); 8 waves, each owns 2 of the 16 output 16x16 tiles.
// ---------------------------------------------------------------------------
__global__ __launch_bounds__(256) void kv_kernel(
    const unsigned short* __restrict__ Kb,
    const unsigned short* __restrict__ Vb,
    float* __restrict__ kv) {
    const int bh = blockIdx.x;
    const int b  = bh >> 4, h = bh & 15;
    __shared__ alignas(16) unsigned short kT[64 * 32];  // [d][s] (= k^T tile)
    __shared__ alignas(16) unsigned short vT[64 * 32];  // [m][s] (= v^T tile)
    const int t  = threadIdx.x;
    const int w  = t >> 5;
    const int dt = w >> 1;         // d-tile 0..3
    const int mt = (w & 1) * 2;    // m-tile 0 or 2 (owns mt, mt+1)

    const v8f vzero = {};
    v8f acc0 = vzero, acc1 = vzero;
    const size_t baseBH = (size_t)b * 8192 * 1024 + (size_t)h * 64;

    const int sl = t >> 3;          // 0..31: local s row
    const int dc = (t & 7) << 3;    // 0..56: d chunk

    for (int s0 = 0; s0 < 8192; s0 += 32) {
        const size_t g = baseBH + (size_t)(s0 + sl) * 1024 + dc;
        alignas(16) unsigned short ku[8], vu[8];
        *reinterpret_cast<v8bf*>(ku) = *reinterpret_cast<const v8bf*>(&Kb[g]);
        *reinterpret_cast<v8bf*>(vu) = *reinterpret_cast<const v8bf*>(&Vb[g]);
#pragma unroll
        for (int e = 0; e < 8; e++) {
            kT[(dc + e) * 32 + sl] = ku[e];
            vT[(dc + e) * 32 + sl] = vu[e];
        }
        __syncthreads();
        v16bf a  = load_frag(&kT[(dt * 16) * 32], 32);
        v16bf b0 = load_frag(&vT[(mt * 16) * 32], 32);
        v16bf b1 = load_frag(&vT[((mt + 1) * 16) * 32], 32);
        acc0 = wmma_bf16(a, b0, acc0);
        acc1 = wmma_bf16(a, b1, acc1);
        __syncthreads();
    }

    const int lane = t & 31;
    const int colL = lane & 15;
    const int rhi  = (lane >> 4) * 8;
    float* out = kv + (size_t)bh * 64 * 64;
#pragma unroll
    for (int r = 0; r < 8; r++) {
        out[(dt * 16 + rhi + r) * 64 + mt * 16 + colL]       = acc0[r];
        out[(dt * 16 + rhi + r) * 64 + (mt + 1) * 16 + colL] = acc1[r];
    }
}

// ksum[b,h,d] = sum_s k[b,s,h,d]
__global__ void ksum_kernel(const unsigned short* __restrict__ Kb,
                            float* __restrict__ ksum) {
    const int bh = blockIdx.x;
    const int b  = bh >> 4, h = bh & 15;
    const int d  = threadIdx.x;  // 64 threads
    float acc = 0.0f;
    size_t base = (size_t)b * 8192 * 1024 + (size_t)h * 64 + d;
    for (int s = 0; s < 8192; s++) acc += bf16_to_f32(Kb[base + (size_t)s * 1024]);
    ksum[bh * 64 + d] = acc;
}

// ---------------------------------------------------------------------------
// attn[b,s,h,m] = (q[b,s,h,:] . kv[b,h,:,m]) / max(q . ksum, eps)
// Block = (s-chunk of 128 rows) x (b,h). 8 waves, 16 rows each, 4 WMMA accums.
// ---------------------------------------------------------------------------
__global__ __launch_bounds__(256) void attn_out_kernel(
    const unsigned short* __restrict__ Qb,
    const float* __restrict__ kv,
    const float* __restrict__ ksum,
    unsigned short* __restrict__ Ob) {
    const int bh    = blockIdx.y;
    const int b     = bh >> 4, h = bh & 15;
    const int s0blk = blockIdx.x * 128;
    __shared__ alignas(16) unsigned short kvT[64 * 64];  // [m][d] bf16
    __shared__ float ks[64];
    const int t = threadIdx.x;

    {   // stage kv^T as bf16 + ksum
        const float* kvp = kv + (size_t)bh * 64 * 64;
        int m  = t >> 2;
        int dc = (t & 3) * 16;
#pragma unroll
        for (int e = 0; e < 16; e++)
            kvT[m * 64 + dc + e] = f32_to_bf16(kvp[(dc + e) * 64 + m]);
        if (t < 64) ks[t] = ksum[bh * 64 + t];
    }
    __syncthreads();

    const int w     = t >> 5;
    const int lane  = t & 31;
    const int srow0 = s0blk + w * 16;
    const size_t qbase = (size_t)b * 8192 * 1024 + (size_t)h * 64;

    // normalizer for row srow0 + (lane&15)
    float norm = 0.0f;
    {
        int row = srow0 + (lane & 15);
        const unsigned short* qrow = Qb + qbase + (size_t)row * 1024;
        for (int d = 0; d < 64; d++) norm += bf16_to_f32(qrow[d]) * ks[d];
    }

    const v8f vz = {};
    v8f acc[4];
#pragma unroll
    for (int j = 0; j < 4; j++) acc[j] = vz;

#pragma unroll
    for (int kt = 0; kt < 64; kt += 32) {
        v16bf a = load_frag(Qb + qbase + (size_t)srow0 * 1024 + kt, 1024);
#pragma unroll
        for (int j = 0; j < 4; j++) {
            v16bf bb = load_frag(&kvT[(j * 16) * 64 + kt], 64);
            acc[j] = wmma_bf16(a, bb, acc[j]);
        }
    }

    const int colL = lane & 15;
    const int rhi  = (lane >> 4) * 8;
#pragma unroll
    for (int j = 0; j < 4; j++) {
#pragma unroll
        for (int r = 0; r < 8; r++) {
            int rl   = rhi + r;
            float nv = __shfl(norm, rl, 32);
            nv = fmaxf(nv, 1e-6f);
            int row = srow0 + rl;
            Ob[((size_t)b * 8192 + row) * 1024 + (size_t)h * 64 + j * 16 + colL] =
                f32_to_bf16(acc[j][r] / nv);
        }
    }
}

// ---------------------------------------------------------------------------
// Launch: X->bf16, W->bf16^T, QKV GEMMs (+elu+1 on Q,K), kv/ksum reduce,
// attention apply, output GEMM -> fp32 d_out.
// ---------------------------------------------------------------------------
extern "C" void kernel_launch(void* const* d_in, const int* in_sizes, int n_in,
                              void* d_out, int out_size, void* d_ws, size_t ws_size,
                              hipStream_t stream) {
    const float* X  = (const float*)d_in[0];
    const float* Wq = (const float*)d_in[1];
    const float* bq = (const float*)d_in[2];
    const float* Wk = (const float*)d_in[3];
    const float* bk = (const float*)d_in[4];
    const float* Wv = (const float*)d_in[5];
    const float* bv = (const float*)d_in[6];
    const float* Wo = (const float*)d_in[7];
    const float* bo = (const float*)d_in[8];

    const size_t MS = 4ULL * 8192ULL;  // B*S = 32768 rows
    const size_t D  = 1024ULL;

    char* ws = (char*)d_ws;
    size_t off = 0;
    auto alloc = [&](size_t bytes) -> void* {
        void* p = ws + off;
        off += (bytes + 255) & ~(size_t)255;
        return p;
    };

    unsigned short* Xb  = (unsigned short*)alloc(MS * D * 2);
    unsigned short* WqT = (unsigned short*)alloc(D * D * 2);
    unsigned short* WkT = (unsigned short*)alloc(D * D * 2);
    unsigned short* WvT = (unsigned short*)alloc(D * D * 2);
    unsigned short* WoT = (unsigned short*)alloc(D * D * 2);
    unsigned short* Qb  = (unsigned short*)alloc(MS * D * 2);
    unsigned short* Kb  = (unsigned short*)alloc(MS * D * 2);
    unsigned short* Vb  = (unsigned short*)alloc(MS * D * 2);
    unsigned short* Ob  = (unsigned short*)alloc(MS * D * 2);
    float* kvbuf  = (float*)alloc(64ULL * 64 * 64 * 4);
    float* ksumb  = (float*)alloc(64ULL * 64 * 4);

    // Precision conversions
    cvt_f32_to_bf16<<<2048, 256, 0, stream>>>(X, Xb, MS * D);
    cvt_transpose_w<<<4096, 256, 0, stream>>>(Wq, WqT);
    cvt_transpose_w<<<4096, 256, 0, stream>>>(Wk, WkT);
    cvt_transpose_w<<<4096, 256, 0, stream>>>(Wv, WvT);
    cvt_transpose_w<<<4096, 256, 0, stream>>>(Wo, WoT);

    // QKV projections (feature map elu+1 fused for Q,K)
    dim3 gg(8 /*N/128*/, 256 /*M/128*/);
    gemm_bf16_kernel<1, false><<<gg, 256, 0, stream>>>(Xb, WqT, bq, Qb, (int)MS);
    gemm_bf16_kernel<1, false><<<gg, 256, 0, stream>>>(Xb, WkT, bk, Kb, (int)MS);
    gemm_bf16_kernel<0, false><<<gg, 256, 0, stream>>>(Xb, WvT, bv, Vb, (int)MS);

    // Per-head kv accumulation + k sums
    kv_kernel<<<64, 256, 0, stream>>>(Kb, Vb, kvbuf);
    ksum_kernel<<<64, 64, 0, stream>>>(Kb, ksumb);

    // Apply attention (q @ kv, normalized)
    dim3 ga(64 /*S/128*/, 64 /*B*H*/);
    attn_out_kernel<<<ga, 256, 0, stream>>>(Qb, kvbuf, ksumb, Ob);

    // Output projection -> fp32
    gemm_bf16_kernel<0, true><<<gg, 256, 0, stream>>>(Ob, WoT, bo, (float*)d_out, (int)MS);
}
